// Weather_forcast_42846593744945
// MI455X (gfx1250) — matrix operand order repaired and checked by reading, not compiled, not run on hardware
//
#include <hip/hip_runtime.h>

#define HN 512   // hidden states
#define HM 256   // timesteps == emission symbols
#define HB 128   // batch

#define INV_LN2 1.4426950408889634f   // 1/ln(2)
#define LN2     0.6931471805599453f

typedef __attribute__((ext_vector_type(16))) _Float16     v16h;
typedef __attribute__((ext_vector_type(8)))  float        v8f;
typedef __attribute__((ext_vector_type(4)))  unsigned int v4u;

#define V_STRIDE 520   // halves (pad: 1040B row stride -> 4-bank shift per row)

// ---------------------------------------------------------------------------
// Kernel 1: column softmax of the transition matrix (axis=0), written as
// linear probabilities directly into the per-lane WMMA B-fragment layout:
//   flat = ((kblk*32 + ntile)*32 + lane)*16 + e
// where B[k][n] = T[n][k], k = kblk*32 + e + 16*(lane>>4), n = ntile*16 + (lane&15)
// ---------------------------------------------------------------------------
__global__ __launch_bounds__(256) void hmm_col_softmax_pack(
        const float* __restrict__ T, _Float16* __restrict__ TtP) {
    __shared__ float red[256];
    const int k = blockIdx.x;
    const int tid = threadIdx.x;
    float x0 = T[(size_t)(tid       ) * HN + k];
    float x1 = T[(size_t)(tid + 256 ) * HN + k];
    red[tid] = fmaxf(x0, x1); __syncthreads();
    for (int s = 128; s > 0; s >>= 1) { if (tid < s) red[tid] = fmaxf(red[tid], red[tid + s]); __syncthreads(); }
    const float mx = red[0]; __syncthreads();
    float e0 = __expf(x0 - mx), e1 = __expf(x1 - mx);
    red[tid] = e0 + e1; __syncthreads();
    for (int s = 128; s > 0; s >>= 1) { if (tid < s) red[tid] += red[tid + s]; __syncthreads(); }
    const float inv = 1.0f / red[0];

    const int kblk = k >> 5, kin = k & 31;
    const int lhalf = kin >> 4, e = kin & 15;
    #pragma unroll
    for (int r = 0; r < 2; ++r) {
        int i = tid + r * 256;                 // state index n
        int lanei = lhalf * 16 + (i & 15);
        int ntile = i >> 4;
        size_t idx = (((size_t)kblk * 32 + ntile) * 32 + lanei) * 16 + e;
        TtP[idx] = (_Float16)((r ? e1 : e0) * inv);
    }
}

// ---------------------------------------------------------------------------
// Kernel 2: row log-softmax of the emission matrix (axis=1), scaled to
// BASE-2 log-probabilities (native domain of v_exp_f32 / v_log_f32)
// ---------------------------------------------------------------------------
__global__ __launch_bounds__(256) void hmm_row_logsoftmax(
        const float* __restrict__ E, float* __restrict__ logE2) {
    __shared__ float red[256];
    const int i = blockIdx.x, tid = threadIdx.x;
    float x = E[(size_t)i * HM + tid];
    red[tid] = x; __syncthreads();
    for (int s = 128; s > 0; s >>= 1) { if (tid < s) red[tid] = fmaxf(red[tid], red[tid + s]); __syncthreads(); }
    float mx = red[0]; __syncthreads();
    red[tid] = __expf(x - mx); __syncthreads();
    for (int s = 128; s > 0; s >>= 1) { if (tid < s) red[tid] += red[tid + s]; __syncthreads(); }
    logE2[(size_t)i * HM + tid] = (x - mx - __logf(red[0])) * INV_LN2;
}

// ---------------------------------------------------------------------------
// Kernel 3: log-softmax of the state priors (base-2)
// ---------------------------------------------------------------------------
__global__ __launch_bounds__(512) void hmm_priors(
        const float* __restrict__ P, float* __restrict__ logP2) {
    __shared__ float red[512];
    const int tid = threadIdx.x;
    float x = P[tid];
    red[tid] = x; __syncthreads();
    for (int s = 256; s > 0; s >>= 1) { if (tid < s) red[tid] = fmaxf(red[tid], red[tid + s]); __syncthreads(); }
    float mx = red[0]; __syncthreads();
    red[tid] = __expf(x - mx); __syncthreads();
    for (int s = 256; s > 0; s >>= 1) { if (tid < s) red[tid] += red[tid + s]; __syncthreads(); }
    logP2[tid] = (x - mx - __logf(red[0])) * INV_LN2;
}

// ---------------------------------------------------------------------------
// Kernel 4: HMM forward recurrence, alpha register-resident in BASE-2 log
// domain.  One workgroup per 16-batch tile (8 WGs), 1024 threads = 32 waves.
// Wave w owns output-state tile [w*16, w*16+16).  Lane layout matches the
// WMMA f32 C/D fragment: lane holds state i = w*16 + (lane&15), batch rows
// b = r + 8*(lane>>4) for acc element r.
// Per step:
//   prefetch: obs[b,t] then logE2[i, obs] gathers (overlap stats + GEMM)
//   stats:    shfl-xor butterflies + 2KB cross-wave reduce -> m_b,
//             V[b,k] = f16(exp2(a-m_b)), log2-sum-exp2 -> out[b,t-1] (*ln2)
//   GEMM:     acc += V_frag x T_frag  (16 x v_wmma_f32_16x16x32_f16, K=512)
//   epilogue: a_new = em2 + log2(acc) + m_b  (bare v_log_f32, registers)
// ---------------------------------------------------------------------------
__global__ __launch_bounds__(1024) void hmm_forward(
        const int*      __restrict__ obs,
        const float*    __restrict__ logP2,
        const float*    __restrict__ logE2,
        const _Float16* __restrict__ TtP,
        float*          __restrict__ out) {
    __shared__ alignas(16) _Float16 V[16 * V_STRIDE];   // exp2(alpha - m_b), f16
    __shared__ float red[16 * 32];                      // per-wave partials
    __shared__ float mb[16];

    const int tid  = threadIdx.x;
    const int wave = tid >> 5, lane = tid & 31;
    const int lm = lane & 15, half = lane >> 4;
    const int bt_base = blockIdx.x * 16;
    const int i = wave * 16 + lm;                       // this lane's state
    const float*    eRow = logE2 + (size_t)i * HM;
    const unsigned* obsU = (const unsigned*)obs;        // values in [0, HM)

    // ---- t = 0: alpha0[b,i] = log2_priors[i] + log2E[i, obs[b,0]]
    float areg[8];
    {
        const float lp = logP2[i];
        #pragma unroll
        for (int r = 0; r < 8; ++r) {
            int b = r + 8 * half;
            areg[r] = lp + eRow[obsU[(size_t)(bt_base + b) * HM]];
        }
    }

    for (int t = 1; t <= HM; ++t) {
        // ---------- prefetch this step's observation indices ----------
        unsigned ob[8];
        if (t < HM) {
            #pragma unroll
            for (int r = 0; r < 8; ++r)
                ob[r] = obsU[(size_t)(bt_base + r + 8 * half) * HM + t];
        }

        // ---------- per-batch-row max over all 512 states ----------
        #pragma unroll
        for (int r = 0; r < 8; ++r) {
            float v = areg[r];
            v = fmaxf(v, __shfl_xor(v, 1, 32));
            v = fmaxf(v, __shfl_xor(v, 2, 32));
            v = fmaxf(v, __shfl_xor(v, 4, 32));
            v = fmaxf(v, __shfl_xor(v, 8, 32));
            if (lm == 0) red[(r + 8 * half) * 32 + wave] = v;
        }
        __syncthreads();
        if (wave < 16) {            // wave w reduces batch row w across 32 waves
            float mm = red[wave * 32 + lane];
            mm = fmaxf(mm, __shfl_xor(mm, 1, 32));
            mm = fmaxf(mm, __shfl_xor(mm, 2, 32));
            mm = fmaxf(mm, __shfl_xor(mm, 4, 32));
            mm = fmaxf(mm, __shfl_xor(mm, 8, 32));
            mm = fmaxf(mm, __shfl_xor(mm, 16, 32));
            if (lane == 0) mb[wave] = mm;
        }
        __syncthreads();

        // ---------- emission gathers (overlap exp pass + GEMM) ----------
        float em[8];
        if (t < HM) {
            #pragma unroll
            for (int r = 0; r < 8; ++r) em[r] = eRow[ob[r]];
        }

        // ---------- V = f16(exp2(a - m_b)); row sums for log2sumexp2 -------
        #pragma unroll
        for (int r = 0; r < 8; ++r) {
            int b = r + 8 * half;
            float e = __builtin_amdgcn_exp2f(areg[r] - mb[b]);   // bare v_exp_f32
            V[b * V_STRIDE + i] = (_Float16)e;
            float v = e;
            v += __shfl_xor(v, 1, 32);
            v += __shfl_xor(v, 2, 32);
            v += __shfl_xor(v, 4, 32);
            v += __shfl_xor(v, 8, 32);
            if (lm == 0) red[b * 32 + wave] = v;
        }
        __syncthreads();
        if (wave < 16) {
            float ss = red[wave * 32 + lane];
            ss += __shfl_xor(ss, 1, 32);
            ss += __shfl_xor(ss, 2, 32);
            ss += __shfl_xor(ss, 4, 32);
            ss += __shfl_xor(ss, 8, 32);
            ss += __shfl_xor(ss, 16, 32);
            if (lane == 0)    // convert base-2 logsumexp back to natural log
                out[(size_t)(bt_base + wave) * HM + (t - 1)] =
                    (mb[wave] + __builtin_amdgcn_logf(ss)) * LN2;
        }
        if (t == HM) break;
        __syncthreads();

        // ---------- GEMM: acc[b, i] = sum_k V[b,k] * T[i,k] ----------
        v8f acc = {};
        const _Float16* vbase = &V[lm * V_STRIDE + 16 * half];
        #pragma unroll 4
        for (int kb = 0; kb < 16; ++kb) {
            union { v4u q[2]; v16h v; } ua, ub;
            // A fragment: two aligned 16B LDS chunks (K = e + 16*half layout)
            const v4u* ap = (const v4u*)(vbase + kb * 32);
            ua.q[0] = ap[0];
            ua.q[1] = ap[1];               // +8 halves
            // B fragment: 32 contiguous bytes per lane from packed T (L2-hot)
            const v4u* bp = (const v4u*)(TtP + (((size_t)kb * 32 + wave) * 32 + lane) * 16);
            ub.q[0] = bp[0];
            ub.q[1] = bp[1];
            acc = __builtin_amdgcn_wmma_f32_16x16x32_f16(
                      /*neg_a=*/false, ua.v, /*neg_b=*/false, ub.v,
                      /*c_mod=*/(short)0, acc, /*reuse_a=*/false, /*reuse_b=*/false);
        }

        // ---------- epilogue: a_new = em2 + log2(acc) + m_b (pure ALU) -----
        #pragma unroll
        for (int r = 0; r < 8; ++r) {
            int b = r + 8 * half;
            areg[r] = em[r] + __builtin_amdgcn_logf(acc[r]) + mb[b];  // bare v_log_f32
        }
        __syncthreads();   // all waves done with V/mb before next iteration
    }
}

// ---------------------------------------------------------------------------
extern "C" void kernel_launch(void* const* d_in, const int* in_sizes, int n_in,
                              void* d_out, int out_size, void* d_ws, size_t ws_size,
                              hipStream_t stream) {
    const int*   obs  = (const int*)  d_in[0];   // [B, M] int32
    const float* upri = (const float*)d_in[1];   // [N]
    const float* utm  = (const float*)d_in[2];   // [N, N]
    const float* uem  = (const float*)d_in[3];   // [N, M]
    float* out = (float*)d_out;                  // [B, M]

    char* ws = (char*)d_ws;
    _Float16* TtP   = (_Float16*)ws;                                    // 512 KB packed f16 T
    float*    logE2 = (float*)(ws + (size_t)HN * HN * 2);               // 512 KB
    float*    logP2 = (float*)(ws + (size_t)HN * HN * 2 + (size_t)HN * HM * 4); // 2 KB

    hmm_col_softmax_pack<<<HN, 256, 0, stream>>>(utm, TtP);
    hmm_row_logsoftmax  <<<HN, 256, 0, stream>>>(uem, logE2);
    hmm_priors          <<<1, 512, 0, stream>>>(upri, logP2);
    hmm_forward<<<HB / 16, 1024, 0, stream>>>(obs, logP2, logE2, TtP, out);
}
